// MultiQueryAttention_5738076307819
// MI455X (gfx1250) — matrix operand order repaired and checked by reading, not compiled, not run on hardware
//
#include <hip/hip_runtime.h>

#define B_   2
#define L_   2048
#define D_   2048
#define H_   16
#define KVH_ 4
#define DK_  128
#define REP_ 4
#define CAP_ 50.0f

typedef __attribute__((ext_vector_type(16))) __bf16 bf16x16;
typedef __attribute__((ext_vector_type(8)))  __bf16 bf16x8;
typedef __attribute__((ext_vector_type(8)))  float  f32x8;

union BF16X16 { bf16x16 v; bf16x8 h[2]; };

__device__ __forceinline__ __bf16 f2bf(float f) {
  union { float f; unsigned u; } a; a.f = f;
  unsigned r = a.u + 0x7FFFu + ((a.u >> 16) & 1u);   // round-to-nearest-even
  union { unsigned short s; __bf16 b; } o; o.s = (unsigned short)(r >> 16);
  return o.b;
}
__device__ __forceinline__ float bf2f(__bf16 b) {
  union { unsigned short s; __bf16 b; } i; i.b = b;
  union { unsigned u; float f; } o; o.u = ((unsigned)i.s) << 16;
  return o.f;
}

// ---- CDNA5 async global->LDS copy (ASYNCcnt-tracked, no VGPR round trip) ----
// Per-lane: LDS[lds_off] = MEM[gaddr], 16 bytes. LDS offset = low 32 bits of
// the generic shared-memory pointer (hardware truncates flat LDS addresses).
__device__ __forceinline__ void async_cp16(const __bf16* g, __bf16* l) {
  unsigned lds_off = (unsigned)(unsigned long long)(const void*)l;
  asm volatile("global_load_async_to_lds_b128 %0, %1, off"
               :: "v"(lds_off), "v"(g) : "memory");
}
__device__ __forceinline__ void wait_async0() {
  asm volatile("s_wait_asynccnt 0x0" ::: "memory");
}

// ---------------------------------------------------------------- converts
__global__ void cvt_f32_to_bf16(const float* __restrict__ src,
                                __bf16* __restrict__ dst, int n) {
  int i = blockIdx.x * blockDim.x + threadIdx.x;
  if (i < n) dst[i] = f2bf(src[i]);
}

// ---------------------------------------------------------------- GEMM
// C[M,N] = A[M,K] (bf16, row-major) x Bw[N,K]^T (bf16, row-major weights)
// 256 threads = 8 waves (4 in M x 2 in N); each wave does a 32x32 patch.
// Double-buffered LDS, filled by async load-to-LDS; one barrier per K-step.
#define GBM 128
#define GBN 64
#define GBK 32
#define APITCH 40   // bf16 elems -> 80B pitch, bank-conflict-free b128 reads
#define BPITCH 40

__global__ __launch_bounds__(256) void gemm_bf16_wmma(
    const __bf16* __restrict__ A, const __bf16* __restrict__ Bw,
    void* __restrict__ Cout, int M, int N, int K, int outF32)
{
  __shared__ __bf16 sA[2][GBM * APITCH];
  __shared__ __bf16 sB[2][GBN * BPITCH];
  const int tid  = threadIdx.x;
  const int wave = tid >> 5, lane = tid & 31;
  const int hl = lane >> 4, l16 = lane & 15;
  const int wm = wave & 3, wn = wave >> 2;
  const int bm0 = blockIdx.x * GBM, bn0 = blockIdx.y * GBN;

  // staging coordinates: 16B per async copy; A = 512 chunks (2/thread), B = 256
  const int ar0 = tid >> 2;
  const int ac8 = (tid & 3) * 8;

  auto stage = [&](int buf, int k0) {
    async_cp16(A  + (size_t)(bm0 + ar0)      * K + k0 + ac8,
               &sA[buf][ar0 * APITCH + ac8]);
    async_cp16(A  + (size_t)(bm0 + ar0 + 64) * K + k0 + ac8,
               &sA[buf][(ar0 + 64) * APITCH + ac8]);
    async_cp16(Bw + (size_t)(bn0 + ar0)      * K + k0 + ac8,
               &sB[buf][ar0 * BPITCH + ac8]);
  };

  f32x8 acc[2][2] = {};

  stage(0, 0);
  wait_async0();
  __syncthreads();

  int cur = 0;
  for (int k0 = 0; k0 < K; k0 += GBK) {
    if (k0 + GBK < K) stage(cur ^ 1, k0 + GBK);   // overlap next tile fetch

    BF16X16 afr[2], bfr[2];
    #pragma unroll
    for (int i = 0; i < 2; ++i) {
      // A fragment (16x32, M x K): half-wave hl reads K {hl*8..} and {16+hl*8..}
      const __bf16* p = &sA[cur][(wm * 32 + i * 16 + l16) * APITCH + hl * 8];
      afr[i].h[0] = *(const bf16x8*)p;
      afr[i].h[1] = *(const bf16x8*)(p + 16);
      // B fragment (32x16, K x N): lane = column n, K {hl*8..} and {16+hl*8..}
      const __bf16* q = &sB[cur][(wn * 32 + i * 16 + l16) * BPITCH + hl * 8];
      bfr[i].h[0] = *(const bf16x8*)q;
      bfr[i].h[1] = *(const bf16x8*)(q + 16);
    }
    #pragma unroll
    for (int i = 0; i < 2; ++i)
      #pragma unroll
      for (int j = 0; j < 2; ++j)
        acc[i][j] = __builtin_amdgcn_wmma_f32_16x16x32_bf16(
            false, afr[i].v, false, bfr[j].v, (short)0, acc[i][j], false, false);

    wait_async0();     // next tile resident in LDS
    __syncthreads();   // everyone done reading cur / writing nxt
    cur ^= 1;
  }

  #pragma unroll
  for (int i = 0; i < 2; ++i)
    #pragma unroll
    for (int j = 0; j < 2; ++j) {
      int n = bn0 + wn * 32 + j * 16 + l16;
      #pragma unroll
      for (int r = 0; r < 8; ++r) {
        int m = bm0 + wm * 32 + i * 16 + hl * 8 + r;
        float v = acc[i][j][r];
        if (outF32) ((float*)Cout)[(size_t)m * N + n] = v;
        else        ((__bf16*)Cout)[(size_t)m * N + n] = f2bf(v);
      }
    }
}

// ---------------------------------------------------------------- RoPE + clip
// src: proj output [B*L, heads*DK]; dst: [B, heads, L, DK] (bf16)
__global__ void rope_clip(const __bf16* __restrict__ src,
                          __bf16* __restrict__ dst, int heads, float scale)
{
  int idx = blockIdx.x * blockDim.x + threadIdx.x;
  int total = B_ * heads * L_ * (DK_ / 2);
  if (idx >= total) return;
  int t = idx;
  int i  = t % (DK_ / 2); t /= (DK_ / 2);
  int l  = t % L_;        t /= L_;
  int hh = t % heads;     int b = t / heads;

  size_t s = (size_t)(b * L_ + l) * (heads * DK_) + (size_t)hh * DK_;
  float x1 = bf2f(src[s + i]);
  float x2 = bf2f(src[s + 64 + i]);
  float inv = powf(10000.0f, -(float)i / 64.0f);
  float f = (float)l * inv;
  float cs = cosf(f), sn = sinf(f);
  float o1 = x1 * cs - x2 * sn;
  float o2 = x1 * sn + x2 * cs;
  o1 = fminf(fmaxf(o1, -CAP_), CAP_) * scale;
  o2 = fminf(fmaxf(o2, -CAP_), CAP_) * scale;
  size_t d = ((size_t)(b * heads + hh) * L_ + l) * DK_;
  dst[d + i]      = f2bf(o1);
  dst[d + 64 + i] = f2bf(o2);
}

// ---------------------------------------------------------------- V transpose
// vp: [B*L, KVH*DK] -> vt: [B, KVH, DK, L]  (so V B-fragments load contiguously)
__global__ void v_transpose(const __bf16* __restrict__ vp,
                            __bf16* __restrict__ vt)
{
  int idx = blockIdx.x * blockDim.x + threadIdx.x;
  int total = B_ * KVH_ * DK_ * L_;
  if (idx >= total) return;
  int t = idx;
  int l  = t % L_;  t /= L_;
  int dk = t % DK_; t /= DK_;
  int kv = t % KVH_; int b = t / KVH_;
  vt[idx] = vp[(size_t)(b * L_ + l) * (KVH_ * DK_) + (size_t)kv * DK_ + dk];
}

// ---------------------------------------------------------------- flash attention
// Qr: [B,H,L,DK] (already clipped and scaled by 1/sqrt(DK))
// Kr: [B,KVH,L,DK] (clipped), Vt: [B,KVH,DK,L], AO: [B*L, D] bf16
__global__ __launch_bounds__(128) void flash_attn(
    const __bf16* __restrict__ Qr, const __bf16* __restrict__ Kr,
    const __bf16* __restrict__ Vt, __bf16* __restrict__ AO)
{
  __shared__ __bf16 sP[4][16 * 40];   // per-wave 16x32 P tile, pitch 40
  const int tile = blockIdx.x, h = blockIdx.y, b = blockIdx.z;
  const int kv = h / REP_;
  const int wave = threadIdx.x >> 5, lane = threadIdx.x & 31;
  const int hl = lane >> 4, l16 = lane & 15;
  const int q0 = tile * 64 + wave * 16;

  const __bf16* Qb = Qr + (size_t)(b * H_ + h)    * L_  * DK_;
  const __bf16* Kb = Kr + (size_t)(b * KVH_ + kv) * L_  * DK_;
  const __bf16* Vb = Vt + (size_t)(b * KVH_ + kv) * DK_ * L_;

  // Q fragments (A-layout, 4 chunks of K=32 over DK=128), kept in registers
  BF16X16 qf[4];
  #pragma unroll
  for (int c = 0; c < 4; ++c) {
    const __bf16* p = Qb + (size_t)(q0 + l16) * DK_ + c * 32 + hl * 8;
    qf[c].h[0] = *(const bf16x8*)p;
    qf[c].h[1] = *(const bf16x8*)(p + 16);
  }

  f32x8 acc[8] = {};
  float mrow[8], lrow[8];
  #pragma unroll
  for (int r = 0; r < 8; ++r) { mrow[r] = -1e30f; lrow[r] = 0.0f; }

  const int kend = (tile + 1) * 64;           // causal bound for this tile
  for (int kb = 0; kb < kend; kb += 32) {
    // prefetch next key block's K rows into near caches (WGP-scope)
    if (kb + 32 < kend) {
      const __bf16* nk = Kb + (size_t)(kb + 32 + l16) * DK_;
      __builtin_prefetch(nk, 0, 3);
      __builtin_prefetch(nk + 16 * DK_, 0, 3);
    }
    // S = Q * K^T for 32 keys (two 16-wide N tiles)
    f32x8 s[2] = {};
    #pragma unroll
    for (int t = 0; t < 2; ++t) {
      int kt = kb + t * 16;
      #pragma unroll
      for (int c = 0; c < 4; ++c) {
        BF16X16 kf;
        const __bf16* p = Kb + (size_t)(kt + l16) * DK_ + c * 32 + hl * 8;
        kf.h[0] = *(const bf16x8*)p;
        kf.h[1] = *(const bf16x8*)(p + 16);
        s[t] = __builtin_amdgcn_wmma_f32_16x16x32_bf16(
            false, qf[c].v, false, kf.v, (short)0, s[t], false, false);
      }
    }
    // causal mask + row-max (reduce across the 16-lane half: N = lane)
    float mc[8];
    #pragma unroll
    for (int r = 0; r < 8; ++r) {
      int row = q0 + hl * 8 + r;
      #pragma unroll
      for (int t = 0; t < 2; ++t) {
        int key = kb + t * 16 + l16;
        if (key > row) s[t][r] = -1e30f;
      }
      float m = fmaxf(s[0][r], s[1][r]);
      #pragma unroll
      for (int off = 1; off < 16; off <<= 1)
        m = fmaxf(m, __shfl_xor(m, off, 32));
      mc[r] = m;
    }
    // online softmax update
    #pragma unroll
    for (int r = 0; r < 8; ++r) {
      float mn = fmaxf(mrow[r], mc[r]);
      float alpha = __expf(mrow[r] - mn);
      float p0 = __expf(s[0][r] - mn);
      float p1 = __expf(s[1][r] - mn);
      s[0][r] = p0; s[1][r] = p1;
      float sum = p0 + p1;
      #pragma unroll
      for (int off = 1; off < 16; off <<= 1)
        sum += __shfl_xor(sum, off, 32);
      lrow[r] = lrow[r] * alpha + sum;
      mrow[r] = mn;
      #pragma unroll
      for (int t8 = 0; t8 < 8; ++t8) acc[t8][r] *= alpha;
    }
    // transpose P (C-layout -> A-layout) via per-wave LDS tile
    #pragma unroll
    for (int t = 0; t < 2; ++t)
      #pragma unroll
      for (int r = 0; r < 8; ++r)
        sP[wave][(hl * 8 + r) * 40 + t * 16 + l16] = f2bf(s[t][r]);
    BF16X16 pf;
    {
      const __bf16* p = &sP[wave][l16 * 40 + hl * 8];
      pf.h[0] = *(const bf16x8*)p;
      pf.h[1] = *(const bf16x8*)(p + 16);
    }
    // O += P * V  (8 dk tiles of 16)
    #pragma unroll
    for (int t8 = 0; t8 < 8; ++t8) {
      BF16X16 vf;
      const __bf16* p = Vb + (size_t)(t8 * 16 + l16) * L_ + kb + hl * 8;
      vf.h[0] = *(const bf16x8*)p;
      vf.h[1] = *(const bf16x8*)(p + 16);
      acc[t8] = __builtin_amdgcn_wmma_f32_16x16x32_bf16(
          false, pf.v, false, vf.v, (short)0, acc[t8], false, false);
    }
  }
  // epilogue: normalize and scatter to [B*L, D] head-merged layout
  float invl[8];
  #pragma unroll
  for (int r = 0; r < 8; ++r) invl[r] = 1.0f / lrow[r];
  #pragma unroll
  for (int t8 = 0; t8 < 8; ++t8) {
    int dk = t8 * 16 + l16;
    #pragma unroll
    for (int r = 0; r < 8; ++r) {
      int row = q0 + hl * 8 + r;
      AO[(size_t)(b * L_ + row) * D_ + h * DK_ + dk] = f2bf(acc[t8][r] * invl[r]);
    }
  }
}

// ---------------------------------------------------------------- launch
extern "C" void kernel_launch(void* const* d_in, const int* in_sizes, int n_in,
                              void* d_out, int out_size, void* d_ws, size_t ws_size,
                              hipStream_t stream)
{
  const float* query = (const float*)d_in[0];
  const float* key_t = (const float*)d_in[1];
  const float* value = (const float*)d_in[2];
  // d_in[3] = causal mask (bool) — causality handled analytically in-kernel
  const float* Wq = (const float*)d_in[4];
  const float* Wk = (const float*)d_in[5];
  const float* Wv = (const float*)d_in[6];
  const float* Wo = (const float*)d_in[7];

  char* ws = (char*)d_ws;
  size_t off = 0;
  auto alloc = [&](size_t bytes) {
    char* p = ws + off; off += (bytes + 255) & ~(size_t)255; return p;
  };
  const int BL = B_ * L_;                               // 4096
  __bf16* qA  = (__bf16*)alloc((size_t)BL * D_ * 2);    // query bf16
  __bf16* kA  = (__bf16*)alloc((size_t)BL * D_ * 2);
  __bf16* vA  = (__bf16*)alloc((size_t)BL * D_ * 2);
  __bf16* wqB = (__bf16*)alloc((size_t)D_ * D_ * 2);
  __bf16* wkB = (__bf16*)alloc((size_t)(D_ / REP_) * D_ * 2);
  __bf16* wvB = (__bf16*)alloc((size_t)(D_ / REP_) * D_ * 2);
  __bf16* woB = (__bf16*)alloc((size_t)D_ * D_ * 2);
  __bf16* QP  = (__bf16*)alloc((size_t)BL * D_ * 2);            // Q proj
  __bf16* KP  = (__bf16*)alloc((size_t)BL * (D_ / REP_) * 2);   // K proj
  __bf16* VP  = (__bf16*)alloc((size_t)BL * (D_ / REP_) * 2);   // V proj
  __bf16* QR  = (__bf16*)alloc((size_t)B_ * H_  * L_ * DK_ * 2);
  __bf16* KR  = (__bf16*)alloc((size_t)B_ * KVH_ * L_ * DK_ * 2);
  __bf16* VT  = (__bf16*)alloc((size_t)B_ * KVH_ * DK_ * L_ * 2);
  __bf16* AO  = (__bf16*)alloc((size_t)BL * D_ * 2);

  int n;
  n = BL * D_;            cvt_f32_to_bf16<<<(n + 255) / 256, 256, 0, stream>>>(query, qA, n);
  n = BL * D_;            cvt_f32_to_bf16<<<(n + 255) / 256, 256, 0, stream>>>(key_t, kA, n);
  n = BL * D_;            cvt_f32_to_bf16<<<(n + 255) / 256, 256, 0, stream>>>(value, vA, n);
  n = D_ * D_;            cvt_f32_to_bf16<<<(n + 255) / 256, 256, 0, stream>>>(Wq, wqB, n);
  n = (D_ / REP_) * D_;   cvt_f32_to_bf16<<<(n + 255) / 256, 256, 0, stream>>>(Wk, wkB, n);
  n = (D_ / REP_) * D_;   cvt_f32_to_bf16<<<(n + 255) / 256, 256, 0, stream>>>(Wv, wvB, n);
  n = D_ * D_;            cvt_f32_to_bf16<<<(n + 255) / 256, 256, 0, stream>>>(Wo, woB, n);

  dim3 gq(BL / GBM, D_ / GBN);
  gemm_bf16_wmma<<<gq, 256, 0, stream>>>(qA, wqB, (void*)QP, BL, D_, D_, 0);
  dim3 gkv(BL / GBM, (D_ / REP_) / GBN);
  gemm_bf16_wmma<<<gkv, 256, 0, stream>>>(kA, wkB, (void*)KP, BL, D_ / REP_, D_, 0);
  gemm_bf16_wmma<<<gkv, 256, 0, stream>>>(vA, wvB, (void*)VP, BL, D_ / REP_, D_, 0);

  const float qscale = 0.08838834764831845f;   // 1/sqrt(DK), folded into Q
  n = B_ * H_ * L_ * (DK_ / 2);
  rope_clip<<<(n + 255) / 256, 256, 0, stream>>>(QP, QR, H_, qscale);
  n = B_ * KVH_ * L_ * (DK_ / 2);
  rope_clip<<<(n + 255) / 256, 256, 0, stream>>>(KP, KR, KVH_, 1.0f);
  n = B_ * KVH_ * DK_ * L_;
  v_transpose<<<(n + 255) / 256, 256, 0, stream>>>(VP, VT);

  dim3 gf(L_ / 64, H_, B_);
  flash_attn<<<gf, 128, 0, stream>>>(QR, KR, VT, AO);

  dim3 go(BL / GBM, D_ / GBN);
  gemm_bf16_wmma<<<go, 256, 0, stream>>>(AO, woB, d_out, BL, D_, D_, 1);
}
